// MaskedMultiHeadAttention_83992380441161
// MI455X (gfx1250) — matrix operand order repaired and checked by reading, not compiled, not run on hardware
//
#include <hip/hip_runtime.h>
#include <hip/hip_bf16.h>

typedef __attribute__((ext_vector_type(16))) _Float16 v16h;
typedef __attribute__((ext_vector_type(8)))  _Float16 v8h;
typedef __attribute__((ext_vector_type(8)))  float    v8f;
typedef __attribute__((ext_vector_type(4)))  float    v4f;
typedef __attribute__((ext_vector_type(4)))  unsigned int u32x4;
typedef __attribute__((ext_vector_type(4)))  int      i32x4;
typedef __attribute__((ext_vector_type(8)))  int      i32x8;

#define WMMA_F16(a, b, c) \
    __builtin_amdgcn_wmma_f32_16x16x32_f16(false, (a), false, (b), (short)0, (c), false, false)

// Problem constants
#define BB   2
#define SS   2048
#define DD   1024
#define HH   16
#define HD   64
#define MM   (BB * SS)               // 4096 flattened rows
#define OUT1 ((size_t)BB * SS * DD)  // first output, floats

// Workspace layout (_Float16 elements), 32 MB total.
#define QOFF ((size_t)0)
#define KOFF ((size_t)4194304)
#define VOFF ((size_t)8388608)   // V transposed: [B,H,64,S]
#define MOFF ((size_t)12582912)  // merged attn output, [B,S,1024] f16

#define LOG2E 1.44269504088896340736f
#define NEG_BIG (-1.0e30f)

// ---------------------------------------------------------------------------
// Tensor Data Mover: 2D f16 tile DMA global->LDS (D# per CDNA5 ISA).
//   tile_d0 = contiguous halfs per line, tile_d1 = #lines,
//   line_stride = halfs between line starts.
// Group0: count=1 | lds_addr | global_addr[56:0] | type=2.
// Group1: data_size=2B, tensor dims (large, no OOB), tile dims, dim0 stride.
// ---------------------------------------------------------------------------
__device__ __forceinline__ void tdm_load_2d_f16(unsigned lds_byte_addr,
                                                const _Float16* gsrc,
                                                unsigned tile_d0, unsigned tile_d1,
                                                unsigned long long line_stride) {
    unsigned long long ga = (unsigned long long)(size_t)(const void*)gsrc;
    u32x4 g0;
    g0[0] = 1u;                                         // count=1, user D#
    g0[1] = lds_byte_addr;                              // LDS dest (bytes)
    g0[2] = (unsigned)ga;                               // global_addr[31:0]
    g0[3] = (unsigned)((ga >> 32) & 0x01FFFFFFu) | (2u << 30);  // [56:32] | type=2
    const unsigned td0 = 1u << 20, td1 = 1u << 20;      // tensor dims: no OOB clip
    i32x8 g1;
    g1[0] = (int)(1u << 16);                            // data_size=1 (2 bytes)
    g1[1] = (int)((td0 & 0xFFFFu) << 16);               // tensor_dim0[15:0]
    g1[2] = (int)(((td0 >> 16) & 0xFFFFu) | ((td1 & 0xFFFFu) << 16));
    g1[3] = (int)(((td1 >> 16) & 0xFFFFu) | ((tile_d0 & 0xFFFFu) << 16));
    g1[4] = (int)(tile_d1 & 0xFFFFu);                   // tile_dim1 (tile_dim2=0)
    g1[5] = (int)(unsigned)(line_stride & 0xFFFFFFFFu); // dim0_stride[31:0]
    g1[6] = (int)(unsigned)((line_stride >> 32) & 0xFFFFu);
    g1[7] = 0;
    i32x4 z4 = {0, 0, 0, 0};
#if __has_include(<hip/amd_detail/amd_gfx1250_TDM.h>)
    i32x8 z8 = {0, 0, 0, 0, 0, 0, 0, 0};
    __builtin_amdgcn_tensor_load_to_lds(g0, g1, z4, z4, z8, 0);
#else
    __builtin_amdgcn_tensor_load_to_lds(g0, g1, z4, z4, 0);
#endif
}

// ---------------------------------------------------------------------------
// Kernel 1: QKV projection.  grid=(64,16,3), block=128 (4 waves).
// Block tile 64x64; each wave owns 16 rows x 64 cols (4 WMMA accumulators),
// reusing one A fragment across 4 WMMAs per K=32 step.  W slab (32x64 f32)
// is cooperatively staged to LDS as f16, transposed for contiguous B-frags.
// ---------------------------------------------------------------------------
__global__ void qkv_proj_kernel(const float* __restrict__ q,
                                const float* __restrict__ k,
                                const float* __restrict__ v,
                                const float* __restrict__ Wq,
                                const float* __restrict__ Wk,
                                const float* __restrict__ Wv,
                                _Float16*    __restrict__ ws) {
    __shared__ alignas(32) _Float16 WldsT[64 * 32];   // [n][kk]

    const int which = blockIdx.z;
    const float* X = (which == 0) ? q  : (which == 1) ? k  : v;
    const float* W = (which == 0) ? Wq : (which == 1) ? Wk : Wv;

    const int tid  = threadIdx.x;
    const int wave = tid >> 5;
    const int lane = tid & 31;
    const int half = lane >> 4;

    const int m0   = blockIdx.x * 64 + wave * 16;
    const int nb   = blockIdx.y * 64;
    const int mrow = m0 + (lane & 15);

    v8f acc[4];
#pragma unroll
    for (int t = 0; t < 4; ++t) acc[t] = (v8f){0.f,0.f,0.f,0.f,0.f,0.f,0.f,0.f};

    for (int kb = 0; kb < DD; kb += 32) {
        // ---- stage W[kb..kb+31][nb..nb+63] -> LDS f16, transposed
        v4f wq4[4];
#pragma unroll
        for (int j = 0; j < 4; ++j) {
            const int qd = tid + j * 128;        // 0..511 quads
            const int kk = qd >> 4;
            const int n4 = (qd & 15) * 4;
            wq4[j] = *(const v4f*)(W + (size_t)(kb + kk) * DD + nb + n4);
        }
        __syncthreads();
#pragma unroll
        for (int j = 0; j < 4; ++j) {
            const int qd = tid + j * 128;
            const int kk = qd >> 4;
            const int n4 = (qd & 15) * 4;
#pragma unroll
            for (int c = 0; c < 4; ++c)
                WldsT[(n4 + c) * 32 + kk] = (_Float16)wq4[j][c];
        }
        __syncthreads();
        if (kb + 32 < DD)  // prefetch next slab (global_prefetch_b8)
            __builtin_prefetch(W + (size_t)(kb + 32 + (tid >> 2)) * DD + nb + (tid & 3) * 16, 0, 1);

        // ---- A fragment (f32 -> f16), reused by 4 WMMAs
        const float* ar = X + (size_t)mrow * DD + kb + half * 16;
        v4f a0 = *(const v4f*)(ar);
        v4f a1 = *(const v4f*)(ar + 4);
        v4f a2 = *(const v4f*)(ar + 8);
        v4f a3 = *(const v4f*)(ar + 12);
        v16h a;
#pragma unroll
        for (int j = 0; j < 4; ++j) {
            a[j]      = (_Float16)a0[j];
            a[4 + j]  = (_Float16)a1[j];
            a[8 + j]  = (_Float16)a2[j];
            a[12 + j] = (_Float16)a3[j];
        }
#pragma unroll
        for (int t = 0; t < 4; ++t) {
            v16h b = *(const v16h*)(WldsT + (t * 16 + (lane & 15)) * 32 + half * 16);
            acc[t] = WMMA_F16(a, b, acc[t]);
        }
    }

    // ---- scatter to head-major f16 layouts
    const int h = blockIdx.y;                 // head size 64 == N block
    if (which < 2) {                          // Q,K -> [B,H,S,64]
        _Float16* out = ws + ((which == 0) ? QOFF : KOFF);
#pragma unroll
        for (int t = 0; t < 4; ++t) {
            const int hd = t * 16 + (lane & 15);
#pragma unroll
            for (int r = 0; r < 8; ++r) {
                const int row = m0 + r + half * 8;
                const int bb  = row >> 11;
                const int s   = row & (SS - 1);
                out[(((size_t)bb * HH + h) * SS + s) * HD + hd] = (_Float16)acc[t][r];
            }
        }
    } else {                                  // V -> transposed [B,H,64,S]
        _Float16* out = ws + VOFF;
        const int bbb = m0 >> 11;
        const int s0  = (m0 & (SS - 1)) + half * 8;
#pragma unroll
        for (int t = 0; t < 4; ++t) {
            const int hd = t * 16 + (lane & 15);
            v8h pk;
#pragma unroll
            for (int r = 0; r < 8; ++r) pk[r] = (_Float16)acc[t][r];
            *(v8h*)(out + (((size_t)bbb * HH + h) * HD + hd) * SS + s0) = pk;
        }
    }
}

// ---------------------------------------------------------------------------
// Kernel 2: causal flash attention.  grid=(S/64, B*H), block=128 (4 waves).
// 4 waves = 4 query 16-row tiles sharing one K/V stream.  Wave0 drives the
// TDM: double-buffered 2D DMA of the K tile (32x64 contiguous lines) and the
// transposed V tile (64 lines of 32 halfs, stride S) into LDS, synchronized
// with s_wait_tensorcnt; barriers publish tiles to the other waves.
// ---------------------------------------------------------------------------
__global__ void flash_attn_kernel(const _Float16* __restrict__ ws_c,
                                  _Float16*       __restrict__ ws,
                                  float*          __restrict__ dout,
                                  const int*      __restrict__ maskedp) {
    __shared__ alignas(32) _Float16 Klds[2][32 * 64];  // [buf][key][hd]
    __shared__ alignas(32) _Float16 Vlds[2][64 * 32];  // [buf][hd][key]
    __shared__ alignas(32) _Float16 Plds[4][16 * 32];  // per-wave P bounce

    const int tid  = threadIdx.x;
    const int wave = tid >> 5;
    const int lane = tid & 31;
    const int half = lane >> 4;
    const int bh   = blockIdx.y;
    const int q0b  = blockIdx.x * 64;
    const int q0   = q0b + wave * 16;
    const int msk  = maskedp[0];

    const _Float16* Qh = ws_c + QOFF + (size_t)bh * SS * HD;
    const _Float16* Kh = ws_c + KOFF + (size_t)bh * SS * HD;
    const _Float16* Vt = ws_c + VOFF + (size_t)bh * HD * SS;

    const unsigned kbyte[2] = {(unsigned)(size_t)(void*)&Klds[0][0],
                               (unsigned)(size_t)(void*)&Klds[1][0]};
    const unsigned vbyte[2] = {(unsigned)(size_t)(void*)&Vlds[0][0],
                               (unsigned)(size_t)(void*)&Vlds[1][0]};

    // Preload Q A-fragments for both K=32 halves of hd=64.
    const _Float16* qr = Qh + (size_t)(q0 + (lane & 15)) * HD;
    v16h qa[2];
#pragma unroll
    for (int s = 0; s < 2; ++s) {
        v8h lo = *(const v8h*)(qr + s * 32 + half * 16);
        v8h hi = *(const v8h*)(qr + s * 32 + half * 16 + 8);
#pragma unroll
        for (int i = 0; i < 8; ++i) { qa[s][i] = lo[i]; qa[s][i + 8] = hi[i]; }
    }

    v8f o0 = {0.f,0.f,0.f,0.f,0.f,0.f,0.f,0.f};
    v8f o1 = o0, o2 = o0, o3 = o0;
    float mrun[8], lrun[8];
#pragma unroll
    for (int r = 0; r < 8; ++r) { mrun[r] = NEG_BIG; lrun[r] = 0.f; }

    const int kend = msk ? (q0b + 64) : SS;   // block-uniform
    const v8f zc = {0.f,0.f,0.f,0.f,0.f,0.f,0.f,0.f};

    if (wave == 0) {                          // prime buffer 0
        tdm_load_2d_f16(kbyte[0], Kh, 64, 32, 64);
        tdm_load_2d_f16(vbyte[0], Vt, 32, 64, SS);
    }

    for (int kbase = 0; kbase < kend; kbase += 32) {
        const int cur = (kbase >> 5) & 1;
        const int nxt = cur ^ 1;
        if (wave == 0) {
            if (kbase + 32 < kend) {          // prefetch next tile, then wait cur
                tdm_load_2d_f16(kbyte[nxt], Kh + (size_t)(kbase + 32) * HD, 64, 32, 64);
                tdm_load_2d_f16(vbyte[nxt], Vt + (kbase + 32), 32, 64, SS);
                __builtin_amdgcn_s_wait_tensorcnt(2);
            } else {
                __builtin_amdgcn_s_wait_tensorcnt(0);
            }
        }
        asm volatile("" ::: "memory");
        __syncthreads();                       // tiles visible to all waves

        const _Float16* Kc = &Klds[cur][0];
        const _Float16* Vc = &Vlds[cur][0];
        const int klo = lane & 15;

        // ---- S = Q K^T for key subtiles [0,16) and [16,32)
        v16h kb00 = *(const v16h*)(Kc + klo * 64 + half * 16);
        v16h kb01 = *(const v16h*)(Kc + klo * 64 + 32 + half * 16);
        v16h kb10 = *(const v16h*)(Kc + (16 + klo) * 64 + half * 16);
        v16h kb11 = *(const v16h*)(Kc + (16 + klo) * 64 + 32 + half * 16);
        v8f s0 = WMMA_F16(qa[0], kb00, zc);
        s0     = WMMA_F16(qa[1], kb01, s0);
        v8f s1 = WMMA_F16(qa[0], kb10, zc);
        s1     = WMMA_F16(qa[1], kb11, s1);

        // ---- scale + causal mask + online softmax (D-fragment aligned)
        const int col0 = kbase + klo;
        const int col1 = col0 + 16;
        float f0[8], f1[8], p0[8], p1[8];
#pragma unroll
        for (int r = 0; r < 8; ++r) {
            const int row = q0 + r + half * 8;
            float x0 = s0[r] * 0.125f;     // 1/sqrt(64)
            float x1 = s1[r] * 0.125f;
            if (msk) {
                x0 = (col0 > row) ? NEG_BIG : x0;
                x1 = (col1 > row) ? NEG_BIG : x1;
            }
            f0[r] = x0; f1[r] = x1;
        }
#pragma unroll
        for (int r = 0; r < 8; ++r) {
            float mx = fmaxf(f0[r], f1[r]);
#pragma unroll
            for (int d = 1; d < 16; d <<= 1)
                mx = fmaxf(mx, __shfl_xor(mx, d, 32));
            const float mnew  = fmaxf(mrun[r], mx);
            const float alpha = exp2f((mrun[r] - mnew) * LOG2E);
            p0[r] = exp2f((f0[r] - mnew) * LOG2E);
            p1[r] = exp2f((f1[r] - mnew) * LOG2E);
            float rs = p0[r] + p1[r];
#pragma unroll
            for (int d = 1; d < 16; d <<= 1)
                rs += __shfl_xor(rs, d, 32);
            lrun[r] = lrun[r] * alpha + rs;
            mrun[r] = mnew;
            o0[r] *= alpha; o1[r] *= alpha; o2[r] *= alpha; o3[r] *= alpha;
        }

        // ---- P: D-layout -> per-wave LDS -> A-layout fragment
        _Float16* Pw = &Plds[wave][0];
#pragma unroll
        for (int r = 0; r < 8; ++r) {
            const int row = r + half * 8;
            Pw[row * 32 + klo]      = (_Float16)p0[r];
            Pw[row * 32 + 16 + klo] = (_Float16)p1[r];
        }
        v8h plo = *(const v8h*)(Pw + klo * 32 + half * 16);
        v8h phi = *(const v8h*)(Pw + klo * 32 + half * 16 + 8);
        v16h pa;
#pragma unroll
        for (int i = 0; i < 8; ++i) { pa[i] = plo[i]; pa[i + 8] = phi[i]; }

        // ---- O += P V (transposed V tile -> contiguous B-fragments)
        v16h vb0 = *(const v16h*)(Vc + (0 * 16 + klo) * 32 + half * 16);
        v16h vb1 = *(const v16h*)(Vc + (1 * 16 + klo) * 32 + half * 16);
        v16h vb2 = *(const v16h*)(Vc + (2 * 16 + klo) * 32 + half * 16);
        v16h vb3 = *(const v16h*)(Vc + (3 * 16 + klo) * 32 + half * 16);
        o0 = WMMA_F16(pa, vb0, o0);
        o1 = WMMA_F16(pa, vb1, o1);
        o2 = WMMA_F16(pa, vb2, o2);
        o3 = WMMA_F16(pa, vb3, o3);

        __syncthreads();                       // reads done before buffer reuse
    }

    // ---- normalize; write attn_out (f32 output #2) + merged f16 (ws)
    float inv[8];
#pragma unroll
    for (int r = 0; r < 8; ++r) inv[r] = 1.0f / lrun[r];

    float*    out2 = dout + OUT1;
    _Float16* Mh   = ws + MOFF;
    const int bbb  = bh >> 4;
    const int h    = bh & 15;

    v8f* ot[4] = {&o0, &o1, &o2, &o3};
#pragma unroll
    for (int t = 0; t < 4; ++t) {
        const int hd = t * 16 + (lane & 15);
#pragma unroll
        for (int r = 0; r < 8; ++r) {
            const int row = q0 + r + half * 8;
            const float val = (*ot[t])[r] * inv[r];
            out2[((size_t)bh * SS + row) * HD + hd] = val;
            Mh[((size_t)bbb * SS + row) * DD + h * HD + hd] = (_Float16)val;
        }
    }
}

// ---------------------------------------------------------------------------
// Kernel 3: output projection merged @ Wo.  grid=(64,16), block=128.
// Same register blocking as kernel 1; A operand already f16.
// ---------------------------------------------------------------------------
__global__ void out_proj_kernel(const _Float16* __restrict__ Mh,
                                const float*    __restrict__ Wo,
                                float*          __restrict__ out) {
    __shared__ alignas(32) _Float16 WldsT[64 * 32];

    const int tid  = threadIdx.x;
    const int wave = tid >> 5;
    const int lane = tid & 31;
    const int half = lane >> 4;

    const int m0   = blockIdx.x * 64 + wave * 16;
    const int nb   = blockIdx.y * 64;
    const int mrow = m0 + (lane & 15);

    v8f acc[4];
#pragma unroll
    for (int t = 0; t < 4; ++t) acc[t] = (v8f){0.f,0.f,0.f,0.f,0.f,0.f,0.f,0.f};

    for (int kb = 0; kb < DD; kb += 32) {
        v4f wq4[4];
#pragma unroll
        for (int j = 0; j < 4; ++j) {
            const int qd = tid + j * 128;
            const int kk = qd >> 4;
            const int n4 = (qd & 15) * 4;
            wq4[j] = *(const v4f*)(Wo + (size_t)(kb + kk) * DD + nb + n4);
        }
        __syncthreads();
#pragma unroll
        for (int j = 0; j < 4; ++j) {
            const int qd = tid + j * 128;
            const int kk = qd >> 4;
            const int n4 = (qd & 15) * 4;
#pragma unroll
            for (int c = 0; c < 4; ++c)
                WldsT[(n4 + c) * 32 + kk] = (_Float16)wq4[j][c];
        }
        __syncthreads();
        if (kb + 32 < DD)
            __builtin_prefetch(Wo + (size_t)(kb + 32 + (tid >> 2)) * DD + nb + (tid & 3) * 16, 0, 1);

        const _Float16* ar = Mh + (size_t)mrow * DD + kb + half * 16;
        v8h lo = *(const v8h*)(ar);
        v8h hi = *(const v8h*)(ar + 8);
        v16h a;
#pragma unroll
        for (int i = 0; i < 8; ++i) { a[i] = lo[i]; a[i + 8] = hi[i]; }

#pragma unroll
        for (int t = 0; t < 4; ++t) {
            v16h b = *(const v16h*)(WldsT + (t * 16 + (lane & 15)) * 32 + half * 16);
            acc[t] = WMMA_F16(a, b, acc[t]);
        }
    }

#pragma unroll
    for (int t = 0; t < 4; ++t) {
        const int ncol = nb + t * 16 + (lane & 15);
#pragma unroll
        for (int r = 0; r < 8; ++r)
            out[(size_t)(m0 + r + half * 8) * DD + ncol] = acc[t][r];
    }
}

// ---------------------------------------------------------------------------
extern "C" void kernel_launch(void* const* d_in, const int* in_sizes, int n_in,
                              void* d_out, int out_size, void* d_ws, size_t ws_size,
                              hipStream_t stream) {
    (void)in_sizes; (void)n_in; (void)out_size; (void)ws_size;

    const float* q  = (const float*)d_in[0];
    const float* k  = (const float*)d_in[1];
    const float* v  = (const float*)d_in[2];
    const float* Wq = (const float*)d_in[3];
    const float* Wk = (const float*)d_in[4];
    const float* Wv = (const float*)d_in[5];
    const float* Wo = (const float*)d_in[6];
    const int* masked = (const int*)d_in[7];

    float*    out = (float*)d_out;
    _Float16* ws  = (_Float16*)d_ws;   // needs 32 MB

    dim3 g1(MM / 64, DD / 64, 3);      // (64, 16, 3)
    qkv_proj_kernel<<<g1, dim3(128), 0, stream>>>(q, k, v, Wq, Wk, Wv, ws);

    dim3 g2(SS / 64, BB * HH);         // (32, 32)
    flash_attn_kernel<<<g2, dim3(128), 0, stream>>>((const _Float16*)ws, ws, out, masked);

    dim3 g3(MM / 64, DD / 64);         // (64, 16)
    out_proj_kernel<<<g3, dim3(128), 0, stream>>>(ws + MOFF, Wo, out);
}
